// PointNetPlusPlus_80616536146693
// MI455X (gfx1250) — matrix operand order repaired and checked by reading, not compiled, not run on hardware
//
#include <hip/hip_runtime.h>

typedef __attribute__((ext_vector_type(16))) _Float16 v16h;
typedef __attribute__((ext_vector_type(8)))  _Float16 v8h;
typedef __attribute__((ext_vector_type(8)))  float    v8f;

#define DEV __device__ __forceinline__

DEV int lane_id() { return (int)(threadIdx.x & 31u); }

// ---------------- WMMA fragment helpers (CDNA5 16x16x32 f16 layouts) ----------------
// A fragment: 16(M) x 32(K) f16, row-major source with leading dim ld.
// lane L: M = L&15, half = L>>4. Per lane the 16 f16 are two contiguous 16B runs:
// cols [half*8 .. half*8+7] and [16+half*8 .. 16+half*8+7]  -> two b128 loads.
DEV v16h load_a_frag(const _Float16* src, int ld) {
  int l = lane_id();
  int m = l & 15, half = l >> 4;
  const _Float16* p = src + m * ld + half * 8;
  union { v16h v; v8h h[2]; } u;
  u.h[0] = *(const v8h*)(p);
  u.h[1] = *(const v8h*)(p + 16);
  return u.v;
}

// B fragment from pre-swizzled fragment-major weights:
// Wf layout = [kt][nt][lane][16 f16]; one aligned 32B load per lane.
DEV v16h load_b_packed(const _Float16* Wf, int kt, int nt, int NT) {
  return ((const v16h*)Wf)[(size_t)(kt * NT + nt) * 32 + lane_id()];
}

DEV v8f wmma32(v16h a, v16h b, v8f c) {
  return __builtin_amdgcn_wmma_f32_16x16x32_f16(false, a, false, b, (short)0, c, false, false);
}

// C/D layout: lane L holds column n = L&15 ; reg r holds row m = r + 8*(L>>4)
DEV void store_tile_relu(_Float16* dstRow0, int ld, int ncol0, v8f c, const float* bias) {
  int l = lane_id(); int n = l & 15, half = l >> 4;
  float bv = bias[ncol0 + n];
#pragma unroll
  for (int r = 0; r < 8; ++r) {
    float v = c[r] + bv; v = v > 0.f ? v : 0.f;
    dstRow0[(r + 8*half) * ld + ncol0 + n] = (_Float16)v;
  }
}

// relu output >= 0, so int-bit atomicMax == float max (buffers initialized to 0).
DEV void maxpool_tile(int* maxb, v8f c, const float* bias, int ncol0) {
  int l = lane_id(); int n = l & 15;
  float bv = bias[ncol0 + n];
  float mx = 0.f;
#pragma unroll
  for (int r = 0; r < 8; ++r) { float v = c[r] + bv; v = v > 0.f ? v : 0.f; mx = fmaxf(mx, v); }
  atomicMax(&maxb[ncol0 + n], __float_as_int(mx));
}

// ---------------- Weight convert: f32 [O][I] -> f16 fragment-major B layout ----------------
// Wf element order: [kt][nt][lane][i], 512 f16 per 32(K)x16(N) tile.
// lane: n_local = lane&15, k = (lane>>4)*8 + i for i<8, 16 + (lane>>4)*8 + (i-8) for i>=8.
__global__ void convert_wt_frag(const float* __restrict__ W, int O, int I, int colOff,
                                int Isub, int Kpad, _Float16* __restrict__ Wf) {
  int e = blockIdx.x * blockDim.x + threadIdx.x;
  int total = Kpad * O;
  if (e >= total) return;
  int NT = O >> 4;
  int i    = e & 15;
  int lane = (e >> 4) & 31;
  int tile = e >> 9;
  int nt = tile % NT;
  int kt = tile / NT;
  int kl = (i < 8) ? (lane >> 4) * 8 + i : 16 + (lane >> 4) * 8 + (i - 8);
  int nl = lane & 15;
  int kg = kt * 32 + kl;
  int ng = nt * 16 + nl;
  float v = (kg < Isub) ? W[(size_t)ng * I + colOff + kg] : 0.f;
  Wf[e] = (_Float16)v;
}

// ---------------- Furthest point sampling: 1 block per batch, points register-resident ----------------
template <int PPT>
__global__ __launch_bounds__(256) void fps_kernel(const float* __restrict__ xyz, int N, int npoint,
                                                  float* __restrict__ new_xyz, int* __restrict__ fidx) {
  int b = blockIdx.x, t = threadIdx.x;
  const float* base = xyz + (size_t)b * N * 3;
  float px[PPT], py[PPT], pz[PPT], mind[PPT];
#pragma unroll
  for (int j = 0; j < PPT; ++j) {
    int p = t + j * 256;
    px[j] = base[3*p]; py[j] = base[3*p+1]; pz[j] = base[3*p+2];
    mind[j] = 1e10f;
  }
  __shared__ float cx, cy, cz;
  __shared__ float redv[256];
  __shared__ int   redi[256];
  if (t == 0) { cx = base[0]; cy = base[1]; cz = base[2]; }
  __syncthreads();
  int far = 0;
  for (int it = 0; it < npoint; ++it) {
    float qx = cx, qy = cy, qz = cz;
    if (t == 0) {
      fidx[b * npoint + it] = far;
      new_xyz[(size_t)(b * npoint + it) * 3 + 0] = qx;
      new_xyz[(size_t)(b * npoint + it) * 3 + 1] = qy;
      new_xyz[(size_t)(b * npoint + it) * 3 + 2] = qz;
    }
    float bestv = -1.f; int besti = 0;
#pragma unroll
    for (int j = 0; j < PPT; ++j) {
      float dx = px[j]-qx, dy = py[j]-qy, dz = pz[j]-qz;
      float d = dx*dx + dy*dy + dz*dz;
      float m = fminf(mind[j], d);
      mind[j] = m;
      if (m > bestv) { bestv = m; besti = t + j * 256; }
    }
    redv[t] = bestv; redi[t] = besti;
    __syncthreads();
    for (int s = 128; s > 0; s >>= 1) {
      if (t < s) {
        float ov = redv[t+s]; int oi = redi[t+s];
        if (ov > redv[t] || (ov == redv[t] && oi < redi[t])) { redv[t] = ov; redi[t] = oi; }
      }
      __syncthreads();
    }
    far = redi[0];                 // everyone reads before owner overwrites centroid
    int owner = far & 255, jj = far >> 8;
    if (t == owner) { cx = px[jj]; cy = py[jj]; cz = pz[jj]; }
    __syncthreads();
  }
}

// ---------------- SA module 1: ballquery(r=0.02, ns=64) + MLP 3->64->128 + maxpool ----------------
__global__ __launch_bounds__(128) void sa1_kernel(
    const float* __restrict__ xyz,        // [32][8192][3]
    const float* __restrict__ newxyz,     // [32][256][3]
    const _Float16* __restrict__ w1f,     // frag-major, Kpad=32, N=64  (NT=4)
    const float* __restrict__ b1,         // [64]
    const _Float16* __restrict__ w2f,     // frag-major, Kpad=64, N=128 (NT=8)
    const float* __restrict__ b2,         // [128]
    _Float16* __restrict__ feat1)         // [32][256][128]
{
  const int N = 8192, NS = 64;
  const float R2 = 0.02f * 0.02f;
  int g = blockIdx.x;                // b*256 + s
  int b = g >> 8;
  int t = threadIdx.x, w = t >> 5, lane = t & 31;
  const float* pbase = xyz + (size_t)b * N * 3;
  float qx = newxyz[(size_t)g*3], qy = newxyz[(size_t)g*3+1], qz = newxyz[(size_t)g*3+2];

  __shared__ int gidx[NS];
  __shared__ int wcnt[4];
  __shared__ __align__(16) _Float16 A[64][32];
  __shared__ __align__(16) _Float16 H[64][64];
  __shared__ int maxb[128];
  maxb[t] = 0;

  // ordered ball-query compaction (first NS smallest indices within radius)
  int total = 0;
  for (int base0 = 0; base0 < N; base0 += 128) {
    if (total >= NS) break;
    int p = base0 + t;
    float dx = pbase[3*p]-qx, dy = pbase[3*p+1]-qy, dz = pbase[3*p+2]-qz;
    bool in = (dx*dx + dy*dy + dz*dz) < R2;
    unsigned m = (unsigned)__ballot(in ? 1 : 0);
    if (lane == 0) wcnt[w] = __popc(m);
    __syncthreads();
    int before = total;
    for (int i = 0; i < w; ++i) before += wcnt[i];
    int pos = before + __popc(m & ((1u << lane) - 1u));
    if (in && pos < NS) gidx[pos] = p;
    total += wcnt[0] + wcnt[1] + wcnt[2] + wcnt[3];
    __syncthreads();
  }
  if (total > NS) total = NS;
  if (total < 1) { if (t == 0) gidx[0] = 0; total = 1; }
  __syncthreads();
  int first = gidx[0];
  for (int e = total + t; e < NS; e += 128) gidx[e] = first;   // pad like reference
  __syncthreads();

  // gather centered coordinates into A (K padded 3 -> 32)
  for (int e = t; e < 64 * 32; e += 128) {
    int row = e >> 5, col = e & 31;
    int pi = gidx[row];
    float v = 0.f;
    if (col == 0) v = pbase[3*pi]   - qx;
    else if (col == 1) v = pbase[3*pi+1] - qy;
    else if (col == 2) v = pbase[3*pi+2] - qz;
    A[row][col] = (_Float16)v;
  }
  __syncthreads();

  // layer1: (64x32)@(32x64) -> relu -> H
  {
    v16h a = load_a_frag(&A[w*16][0], 32);
    for (int n = 0; n < 4; ++n) {
      v8f c = {};
      c = wmma32(a, load_b_packed(w1f, 0, n, 4), c);
      store_tile_relu(&H[w*16][0], 64, n*16, c, b1);
    }
  }
  __syncthreads();

  // layer2: (64x64)@(64x128) -> relu -> max over 64 samples
  {
    v16h a0 = load_a_frag(&H[w*16][0],  64);
    v16h a1 = load_a_frag(&H[w*16][32], 64);
    for (int n = 0; n < 8; ++n) {
      v8f c = {};
      c = wmma32(a0, load_b_packed(w2f, 0, n, 8), c);
      c = wmma32(a1, load_b_packed(w2f, 1, n, 8), c);
      maxpool_tile(maxb, c, b2, n*16);
    }
  }
  __syncthreads();
  feat1[(size_t)g * 128 + t] = (_Float16)__int_as_float(maxb[t]);
}

// ---------------- SA module 2: ballquery(r=0.04, ns=128) + MLP 131->128->256 + maxpool ----------------
__global__ __launch_bounds__(128) void sa2_kernel(
    const float* __restrict__ xyz2,       // [32][256][3]
    const float* __restrict__ newxyz,     // [32][64][3]
    const _Float16* __restrict__ feat1,   // [32][256][128]
    const _Float16* __restrict__ w1xf,    // frag-major, Kpad=32,  N=128 (NT=8)
    const _Float16* __restrict__ w1ff,    // frag-major, Kpad=128, N=128 (NT=8)
    const float* __restrict__ b1,         // [128]
    const _Float16* __restrict__ w2f,     // frag-major, Kpad=128, N=256 (NT=16)
    const float* __restrict__ b2,         // [256]
    _Float16* __restrict__ feat2)         // [32][64][256]
{
  const int N = 256, NS = 128;
  const float R2 = 0.04f * 0.04f;
  int g = blockIdx.x;                 // b*64 + s
  int b = g >> 6;
  int t = threadIdx.x, w = t >> 5, lane = t & 31;
  const float* pbase = xyz2 + (size_t)b * N * 3;
  const _Float16* fbase = feat1 + (size_t)b * 256 * 128;
  float qx = newxyz[(size_t)g*3], qy = newxyz[(size_t)g*3+1], qz = newxyz[(size_t)g*3+2];

  __shared__ int gidx[NS];
  __shared__ int wcnt[4];
  __shared__ __align__(16) _Float16 Ax[64][32];
  __shared__ __align__(16) _Float16 Af[64][128];
  __shared__ __align__(16) _Float16 H[64][128];
  __shared__ int maxb[256];
  maxb[t] = 0; maxb[t + 128] = 0;

  int total = 0;
  for (int base0 = 0; base0 < N; base0 += 128) {
    if (total >= NS) break;
    int p = base0 + t;
    float dx = pbase[3*p]-qx, dy = pbase[3*p+1]-qy, dz = pbase[3*p+2]-qz;
    bool in = (dx*dx + dy*dy + dz*dz) < R2;
    unsigned m = (unsigned)__ballot(in ? 1 : 0);
    if (lane == 0) wcnt[w] = __popc(m);
    __syncthreads();
    int before = total;
    for (int i = 0; i < w; ++i) before += wcnt[i];
    int pos = before + __popc(m & ((1u << lane) - 1u));
    if (in && pos < NS) gidx[pos] = p;
    total += wcnt[0] + wcnt[1] + wcnt[2] + wcnt[3];
    __syncthreads();
  }
  if (total > NS) total = NS;
  if (total < 1) { if (t == 0) gidx[0] = 0; total = 1; }
  __syncthreads();
  int first = gidx[0];
  for (int e = total + t; e < NS; e += 128) gidx[e] = first;

  for (int ch = 0; ch < 2; ++ch) {        // 64-row chunks of the 128 samples
    __syncthreads();
    for (int e = t; e < 64 * 32; e += 128) {
      int row = e >> 5, col = e & 31;
      int pi = gidx[ch*64 + row];
      float v = 0.f;
      if (col == 0) v = pbase[3*pi]   - qx;
      else if (col == 1) v = pbase[3*pi+1] - qy;
      else if (col == 2) v = pbase[3*pi+2] - qz;
      Ax[row][col] = (_Float16)v;
    }
    // vectorized feature gather: 16 x uint4 per 128-f16 row
    for (int e = t; e < 64 * 16; e += 128) {
      int row = e >> 4, q = e & 15;
      ((uint4*)&Af[row][0])[q] =
          ((const uint4*)(fbase + (size_t)gidx[ch*64 + row] * 128))[q];
    }
    __syncthreads();
    // layer1: split-K (xyz pad32 + feat 128)
    {
      v16h ax  = load_a_frag(&Ax[w*16][0], 32);
      v16h af0 = load_a_frag(&Af[w*16][0],  128);
      v16h af1 = load_a_frag(&Af[w*16][32], 128);
      v16h af2 = load_a_frag(&Af[w*16][64], 128);
      v16h af3 = load_a_frag(&Af[w*16][96], 128);
      for (int n = 0; n < 8; ++n) {
        v8f c = {};
        c = wmma32(ax,  load_b_packed(w1xf, 0, n, 8), c);
        c = wmma32(af0, load_b_packed(w1ff, 0, n, 8), c);
        c = wmma32(af1, load_b_packed(w1ff, 1, n, 8), c);
        c = wmma32(af2, load_b_packed(w1ff, 2, n, 8), c);
        c = wmma32(af3, load_b_packed(w1ff, 3, n, 8), c);
        store_tile_relu(&H[w*16][0], 128, n*16, c, b1);
      }
    }
    __syncthreads();
    // layer2 + maxpool
    {
      v16h h0 = load_a_frag(&H[w*16][0],  128);
      v16h h1 = load_a_frag(&H[w*16][32], 128);
      v16h h2 = load_a_frag(&H[w*16][64], 128);
      v16h h3 = load_a_frag(&H[w*16][96], 128);
      for (int n = 0; n < 16; ++n) {
        v8f c = {};
        c = wmma32(h0, load_b_packed(w2f, 0, n, 16), c);
        c = wmma32(h1, load_b_packed(w2f, 1, n, 16), c);
        c = wmma32(h2, load_b_packed(w2f, 2, n, 16), c);
        c = wmma32(h3, load_b_packed(w2f, 3, n, 16), c);
        maxpool_tile(maxb, c, b2, n*16);
      }
    }
  }
  __syncthreads();
  feat2[(size_t)g * 256 + t]       = (_Float16)__int_as_float(maxb[t]);
  feat2[(size_t)g * 256 + 128 + t] = (_Float16)__int_as_float(maxb[128 + t]);
}

// ---------------- group_all: MLP 259->256->512 + maxpool over 64 points ----------------
__global__ __launch_bounds__(128) void sa3_kernel(
    const float* __restrict__ xyz2,       // [32][64][3] (= new_xyz2, NOT centered)
    const _Float16* __restrict__ feat2,   // [32][64][256]
    const _Float16* __restrict__ w1xf,    // frag-major, Kpad=32,  N=256 (NT=16)
    const _Float16* __restrict__ w1ff,    // frag-major, Kpad=256, N=256 (NT=16)
    const float* __restrict__ b1,         // [256]
    const _Float16* __restrict__ w2f,     // frag-major, Kpad=256, N=512 (NT=32)
    const float* __restrict__ b2,         // [512]
    _Float16* __restrict__ feat3)         // [32][512]
{
  int b = blockIdx.x;
  int t = threadIdx.x, w = t >> 5;
  __shared__ __align__(16) _Float16 Ax[32][32];
  __shared__ __align__(16) _Float16 Af[32][256];
  __shared__ __align__(16) _Float16 H[32][256];
  __shared__ int maxb[512];
  for (int e = t; e < 512; e += 128) maxb[e] = 0;
  const float* pb = xyz2 + (size_t)b * 64 * 3;
  const _Float16* fb = feat2 + (size_t)b * 64 * 256;

  for (int ch = 0; ch < 2; ++ch) {        // 32-row chunks of the 64 points
    __syncthreads();
    for (int e = t; e < 32 * 32; e += 128) {
      int row = e >> 5, col = e & 31;
      int pi = ch*32 + row;
      float v = 0.f;
      if (col == 0) v = pb[3*pi];
      else if (col == 1) v = pb[3*pi+1];
      else if (col == 2) v = pb[3*pi+2];
      Ax[row][col] = (_Float16)v;
    }
    // vectorized copy: 32 x uint4 per 256-f16 row
    for (int e = t; e < 32 * 32; e += 128) {
      int row = e >> 5, q = e & 31;
      ((uint4*)&Af[row][0])[q] =
          ((const uint4*)(fb + (size_t)(ch*32 + row) * 256))[q];
    }
    __syncthreads();
    // layer1: 2 x 16 tiles, round-robin over waves
    for (int tid = w; tid < 32; tid += 4) {
      int m = tid >> 4, n = tid & 15;
      v8f c = {};
      c = wmma32(load_a_frag(&Ax[m*16][0], 32), load_b_packed(w1xf, 0, n, 16), c);
      for (int ks = 0; ks < 8; ++ks)
        c = wmma32(load_a_frag(&Af[m*16][ks*32], 256),
                   load_b_packed(w1ff, ks, n, 16), c);
      store_tile_relu(&H[m*16][0], 256, n*16, c, b1);
    }
    __syncthreads();
    // layer2: 2 x 32 tiles
    for (int tid = w; tid < 64; tid += 4) {
      int m = tid >> 5, n = tid & 31;
      v8f c = {};
      for (int ks = 0; ks < 8; ++ks)
        c = wmma32(load_a_frag(&H[m*16][ks*32], 256),
                   load_b_packed(w2f, ks, n, 32), c);
      maxpool_tile(maxb, c, b2, n*16);
    }
  }
  __syncthreads();
  for (int e = t; e < 512; e += 128)
    feat3[(size_t)b * 512 + e] = (_Float16)__int_as_float(maxb[e]);
}

// ---------------- head: generic M=32 GEMM, one 16x16 tile per wave ----------------
__global__ __launch_bounds__(128) void head_gemm(
    const _Float16* __restrict__ A, int K,    // [32][K] f16
    const _Float16* __restrict__ Wf, int Nn,  // frag-major [K/32][Nn/16][lane][16]
    const float* __restrict__ bias, int do_relu,
    _Float16* __restrict__ out16,             // [32][Nn] or null
    float* __restrict__ out32)                // [32][Nn] or null
{
  int w = (int)(threadIdx.x >> 5);
  int tile = blockIdx.x * 4 + w;
  int ntiles = 2 * (Nn >> 4);
  if (tile >= ntiles) return;
  int m = tile & 1, n = tile >> 1;
  int NT = Nn >> 4;
  v8f c = {};
  const _Float16* arow = A + (size_t)m * 16 * K;
  for (int kt = 0; kt < (K >> 5); ++kt)
    c = wmma32(load_a_frag(arow + kt * 32, K), load_b_packed(Wf, kt, n, NT), c);
  int l = lane_id(); int nn = l & 15, half = l >> 4;
  int col = n*16 + nn;
  float bv = bias[col];
#pragma unroll
  for (int r = 0; r < 8; ++r) {
    int row = m*16 + r + 8*half;
    float v = c[r] + bv;
    if (do_relu) v = v > 0.f ? v : 0.f;
    if (out16) out16[(size_t)row * Nn + col] = (_Float16)v;
    if (out32) out32[(size_t)row * Nn + col] = v;
  }
}

// ---------------- launch ----------------
extern "C" void kernel_launch(void* const* d_in, const int* in_sizes, int n_in,
                              void* d_out, int out_size, void* d_ws, size_t ws_size,
                              hipStream_t stream) {
  (void)in_sizes; (void)n_in; (void)out_size; (void)ws_size;
  const float* pc  = (const float*)d_in[0];
  const float* w11 = (const float*)d_in[1];  const float* b11 = (const float*)d_in[2];
  const float* w12 = (const float*)d_in[3];  const float* b12 = (const float*)d_in[4];
  const float* w21 = (const float*)d_in[5];  const float* b21 = (const float*)d_in[6];
  const float* w22 = (const float*)d_in[7];  const float* b22 = (const float*)d_in[8];
  const float* w31 = (const float*)d_in[9];  const float* b31 = (const float*)d_in[10];
  const float* w32 = (const float*)d_in[11]; const float* b32 = (const float*)d_in[12];
  const float* hw1 = (const float*)d_in[13]; const float* hb1 = (const float*)d_in[14];
  const float* hw2 = (const float*)d_in[15]; const float* hb2 = (const float*)d_in[16];
  const float* hw3 = (const float*)d_in[17]; const float* hb3 = (const float*)d_in[18];

  char* ws = (char*)d_ws;
  size_t off = 0;
  auto alloc = [&](size_t bytes) -> char* {
    char* p = ws + off;
    off = (off + bytes + 255) & ~(size_t)255;
    return p;
  };
  _Float16* Wt11  = (_Float16*)alloc((size_t)32*64*2);
  _Float16* Wt12  = (_Float16*)alloc((size_t)64*128*2);
  _Float16* Wt21x = (_Float16*)alloc((size_t)32*128*2);
  _Float16* Wt21f = (_Float16*)alloc((size_t)128*128*2);
  _Float16* Wt22  = (_Float16*)alloc((size_t)128*256*2);
  _Float16* Wt31x = (_Float16*)alloc((size_t)32*256*2);
  _Float16* Wt31f = (_Float16*)alloc((size_t)256*256*2);
  _Float16* Wt32  = (_Float16*)alloc((size_t)256*512*2);
  _Float16* WtH1  = (_Float16*)alloc((size_t)512*1024*2);
  _Float16* WtH2  = (_Float16*)alloc((size_t)1024*1024*2);
  _Float16* WtH3  = (_Float16*)alloc((size_t)1024*512*2);
  int*      fidx1 = (int*)alloc((size_t)32*256*4);
  float*    nx1   = (float*)alloc((size_t)32*256*3*4);
  int*      fidx2 = (int*)alloc((size_t)32*64*4);
  float*    nx2   = (float*)alloc((size_t)32*64*3*4);
  _Float16* feat1 = (_Float16*)alloc((size_t)32*256*128*2);
  _Float16* feat2 = (_Float16*)alloc((size_t)32*64*256*2);
  _Float16* feat3 = (_Float16*)alloc((size_t)32*512*2);
  _Float16* h1    = (_Float16*)alloc((size_t)32*1024*2);
  _Float16* h2    = (_Float16*)alloc((size_t)32*1024*2);

  auto cw = [&](const float* W, int O, int I, int colOff, int Isub, int Kpad, _Float16* Wf) {
    int total = Kpad * O;
    convert_wt_frag<<<(total + 255) / 256, 256, 0, stream>>>(W, O, I, colOff, Isub, Kpad, Wf);
  };
  cw(w11,   64,    3, 0,    3,   32, Wt11);
  cw(w12,  128,   64, 0,   64,   64, Wt12);
  cw(w21,  128,  131, 0,    3,   32, Wt21x);
  cw(w21,  128,  131, 3,  128,  128, Wt21f);
  cw(w22,  256,  128, 0,  128,  128, Wt22);
  cw(w31,  256,  259, 0,    3,   32, Wt31x);
  cw(w31,  256,  259, 3,  256,  256, Wt31f);
  cw(w32,  512,  256, 0,  256,  256, Wt32);
  cw(hw1, 1024,  512, 0,  512,  512, WtH1);
  cw(hw2, 1024, 1024, 0, 1024, 1024, WtH2);
  cw(hw3,  512, 1024, 0, 1024, 1024, WtH3);

  fps_kernel<32><<<32, 256, 0, stream>>>(pc, 8192, 256, nx1, fidx1);
  sa1_kernel<<<32 * 256, 128, 0, stream>>>(pc, nx1, Wt11, b11, Wt12, b12, feat1);
  fps_kernel<1><<<32, 256, 0, stream>>>(nx1, 256, 64, nx2, fidx2);
  sa2_kernel<<<32 * 64, 128, 0, stream>>>(nx1, nx2, feat1, Wt21x, Wt21f, b21, Wt22, b22, feat2);
  sa3_kernel<<<32, 128, 0, stream>>>(nx2, feat2, Wt31x, Wt31f, b31, Wt32, b32, feat3);

  head_gemm<<<32, 128, 0, stream>>>(feat3,  512, WtH1, 1024, hb1, 1, h1, nullptr);
  head_gemm<<<32, 128, 0, stream>>>(h1,    1024, WtH2, 1024, hb2, 1, h2, nullptr);
  head_gemm<<<16, 128, 0, stream>>>(h2,    1024, WtH3,  512, hb3, 0, nullptr, (float*)d_out);
}
